// TGCNLayer_16896401342874
// MI455X (gfx1250) — compile-verified, matrix operand order checked by reading, not run on hardware
//
#include <hip/hip_runtime.h>
#include <hip/hip_bf16.h>

// ---------------------------------------------------------------------------
// TGCN layer for MI455X (gfx1250, wave32, WMMA + TDM).
// Compute-bound (~64 GFLOP) -> v_wmma_f32_16x16x32_bf16 with fp32 accum.
// Wave blocking: each wave owns 2 N-tiles x 4 M-tiles so every B fragment is
// loaded once per block and reused 4x from registers. The conv batch-window
// staging into LDS is done by the Tensor Data Mover (tensor_load_to_lds,
// tracked by TENSORcnt) instead of VALU loads.
// ---------------------------------------------------------------------------

typedef __bf16 bf16;
typedef bf16  v16bf __attribute__((ext_vector_type(16)));
typedef bf16  v8bf  __attribute__((ext_vector_type(8)));
typedef bf16  v4bf  __attribute__((ext_vector_type(4)));
typedef float v8f   __attribute__((ext_vector_type(8)));
typedef unsigned int u32x4 __attribute__((ext_vector_type(4)));
typedef unsigned int u32x8 __attribute__((ext_vector_type(8)));

#define BQ   2048
#define NN   64
#define CIND 128
#define CO   256
#define EE   256
#define EPSV 1e-5f

// A-matrix 16x32 bf16 frag (row-major source): lane ln holds row M=ln;
// half (lane>=16) selects K offsets {h*8..h*8+7, 16+h*8..16+h*8+7}.
__device__ __forceinline__ v16bf load_a_frag(const bf16* row_kblk, int half) {
    v8bf lo = *(const v8bf*)(row_kblk + half * 8);
    v8bf hi = *(const v8bf*)(row_kblk + 16 + half * 8);
    return __builtin_shufflevector(lo, hi, 0,1,2,3,4,5,6,7,8,9,10,11,12,13,14,15);
}

__device__ __forceinline__ v8f wmma_bf16(v16bf a, v16bf b, v8f c) {
    return __builtin_amdgcn_wmma_f32_16x16x32_bf16(false, a, false, b,
                                                   (short)0, c, false, false);
}

// packed B-frag layout: elem index = (kb<<13) + (n<<5) + (half<<4) + i
// stores B[kb*32 + k_of(i,half)][n]; one 32B vector load per fragment.
__device__ __forceinline__ int k_of(int i, int h) {
    return (i & 7) + ((i >> 3) << 4) + (h << 3);
}

// --- kernel A: graph matrix + weight pre-swizzle ---------------------------
__global__ __launch_bounds__(256) void k_setup(const float* __restrict__ W,
                                               const float* __restrict__ w_conv,
                                               const int* __restrict__ ei,
                                               bf16* __restrict__ wA,
                                               bf16* __restrict__ wW,
                                               bf16* __restrict__ wWk) {
    __shared__ int   sdeg[NN];
    __shared__ float sdinv[NN];
    __shared__ float sAf[NN * NN];
    const int t = threadIdx.x;
    if (t < NN) sdeg[t] = 1;                       // self loop
    __syncthreads();
    if (t < EE) atomicAdd(&sdeg[ei[EE + t]], 1);   // dst degree (int: deterministic)
    __syncthreads();
    if (t < NN) sdinv[t] = rsqrtf((float)sdeg[t]);
    __syncthreads();
    for (int i = t; i < NN * NN; i += 256) sAf[i] = 0.f;
    __syncthreads();
    if (t < EE) {
        int s = ei[t], d = ei[EE + t];
        atomicAdd(&sAf[d * NN + s], sdinv[s] * sdinv[d]); // identical addends per addr
    }
    if (t < NN) atomicAdd(&sAf[t * NN + t], sdinv[t] * sdinv[t]);
    __syncthreads();
    for (int i = t; i < NN * NN; i += 256) wA[i] = (bf16)sAf[i];

    // pack W [128,256] into B-frag layout (4 kb blocks)
    for (int idx = t; idx < 4 * CO * 32; idx += 256) {
        int i = idx & 15, h = (idx >> 4) & 1, n = (idx >> 5) & 255, kb = idx >> 13;
        wW[idx] = (bf16)W[(kb * 32 + k_of(i, h)) * CO + n];
    }
    // pack w_conv [co][ci][k] into 3 B-frag matrices Wk[ci][co] (8 kb blocks each)
    for (int idx = t; idx < 3 * 8 * CO * 32; idx += 256) {
        int k = idx >> 16, r = idx & 65535;
        int i = r & 15, h = (r >> 4) & 1, n = (r >> 5) & 255, kb = r >> 13;
        int ci = kb * 32 + k_of(i, h);
        wWk[idx] = (bf16)w_conv[n * (CO * 3) + ci * 3 + k];
    }
}

// --- kernel B: per-batch x_gcn = Ahat @ (x[b] @ W) + bias ------------------
__global__ __launch_bounds__(256) void k_gcn(const float* __restrict__ x,
                                             const float* __restrict__ b_gcn,
                                             const bf16* __restrict__ wA,
                                             const bf16* __restrict__ wW,
                                             bf16* __restrict__ xg) {
    __shared__ alignas(32) bf16 sx[NN * CIND];   // 16 KB
    __shared__ alignas(32) bf16 sA[NN * NN];     //  8 KB
    __shared__ alignas(32) bf16 sxw[2 * CO * 32];// 32 KB (B-frag packed xw)
    const int b = blockIdx.x;
    const int t = threadIdx.x;
    // vectorized fp32 -> bf16 staging of x[b]
    {
        const float4* xv = (const float4*)(x + (size_t)b * (NN * CIND));
        for (int i = t; i < (NN * CIND) / 4; i += 256) {
            float4 v = xv[i];
            v4bf w; w[0] = (bf16)v.x; w[1] = (bf16)v.y; w[2] = (bf16)v.z; w[3] = (bf16)v.w;
            *(v4bf*)(sx + i * 4) = w;
        }
        const uint4* av = (const uint4*)wA;
        for (int i = t; i < (NN * NN * 2) / 16; i += 256) ((uint4*)sA)[i] = av[i];
    }
    __syncthreads();

    const int lane = t & 31, wave = t >> 5;
    const int half = lane >> 4, ln = lane & 15;
    const int nt0 = wave * 2;                 // this wave owns n-tiles {nt0, nt0+1}
    const int co0 = nt0 * 16 + ln;
    const int co1 = co0 + 16;

    v8f acc[8];                                // acc[mt*2 + p]
    for (int j = 0; j < 8; ++j) for (int r = 0; r < 8; ++r) acc[j][r] = 0.f;

    // GEMM1: xw = x @ W   (M=64, N=256, K=128); B frags loaded once, reused 4x
    for (int kb = 0; kb < 4; ++kb) {
        const v16bf bm0 = *(const v16bf*)(wW + (kb << 13) + (co0 << 5) + (half << 4));
        const v16bf bm1 = *(const v16bf*)(wW + (kb << 13) + (co1 << 5) + (half << 4));
        #pragma unroll
        for (int mt = 0; mt < 4; ++mt) {
            v16bf a = load_a_frag(sx + (mt * 16 + ln) * CIND + kb * 32, half);
            acc[mt * 2 + 0] = wmma_bf16(a, bm0, acc[mt * 2 + 0]);
            acc[mt * 2 + 1] = wmma_bf16(a, bm1, acc[mt * 2 + 1]);
        }
    }
    // scatter C-layout result into B-frag packed LDS for GEMM2
    #pragma unroll
    for (int mt = 0; mt < 4; ++mt) {
        #pragma unroll
        for (int p = 0; p < 2; ++p) {
            const int co = (p == 0) ? co0 : co1;
            #pragma unroll
            for (int r = 0; r < 8; ++r) {
                const int M = mt * 16 + half * 8 + r;
                const int kb2 = M >> 5, kk = M & 31;
                const int h2 = (kk >> 3) & 1;
                const int i2 = (kk & 7) + ((kk >> 4) << 3);
                sxw[(kb2 << 13) + (co << 5) + (h2 << 4) + i2] = (bf16)acc[mt * 2 + p][r];
            }
        }
    }
    __syncthreads();

    // GEMM2: Ahat @ xw   (M=64, N=256, K=64)
    for (int j = 0; j < 8; ++j) for (int r = 0; r < 8; ++r) acc[j][r] = 0.f;
    for (int kb = 0; kb < 2; ++kb) {
        const v16bf bm0 = *(const v16bf*)(sxw + (kb << 13) + (co0 << 5) + (half << 4));
        const v16bf bm1 = *(const v16bf*)(sxw + (kb << 13) + (co1 << 5) + (half << 4));
        #pragma unroll
        for (int mt = 0; mt < 4; ++mt) {
            v16bf a = load_a_frag(sA + (mt * 16 + ln) * NN + kb * 32, half);
            acc[mt * 2 + 0] = wmma_bf16(a, bm0, acc[mt * 2 + 0]);
            acc[mt * 2 + 1] = wmma_bf16(a, bm1, acc[mt * 2 + 1]);
        }
    }
    const float bias0 = b_gcn[co0];
    const float bias1 = b_gcn[co1];
    #pragma unroll
    for (int mt = 0; mt < 4; ++mt) {
        #pragma unroll
        for (int r = 0; r < 8; ++r) {
            const int node = mt * 16 + half * 8 + r;
            const size_t base = (size_t)node * (BQ * CO) + (size_t)b * CO;
            xg[base + co0] = (bf16)(acc[mt * 2 + 0][r] + bias0);
            xg[base + co1] = (bf16)(acc[mt * 2 + 1][r] + bias1);
        }
    }
}

// --- kernel C: temporal conv over batch axis per node ----------------------
// Batch window staged into LDS by the Tensor Data Mover (one 2D-tile D#).
__global__ __launch_bounds__(256) void k_conv(const bf16* __restrict__ xg,
                                              const bf16* __restrict__ wWk,
                                              const float* __restrict__ b_conv,
                                              float* __restrict__ out) {
    __shared__ alignas(32) bf16 sX[66 * CO];     // 33 KB zero-padded batch window
    const int b0 = blockIdx.x * 64;
    const int node = blockIdx.y;
    const int t = threadIdx.x;
    const size_t nbase = (size_t)node * (BQ * CO);

    if (t < 32) {                                 // wave 0: pad rows + issue TDM
        // zero the out-of-range pad row(s) (b = -1 / b = BQ) in LDS
        v8bf z;
        #pragma unroll
        for (int e = 0; e < 8; ++e) z[e] = (bf16)0.f;
        if (b0 == 0)       *(v8bf*)(sX + t * 8) = z;            // window row 0
        if (b0 + 64 == BQ) *(v8bf*)(sX + 65 * CO + t * 8) = z;  // window row 65

        // clamp tile rows to the tensor; dest offset keeps window alignment
        const int start = (b0 == 0) ? 0 : (b0 - 1);
        const int endr  = (b0 + 65 > BQ) ? BQ : (b0 + 65);
        const unsigned nrows = (unsigned)(endr - start);
        const unsigned long long gaddr =
            (unsigned long long)(uintptr_t)(xg + nbase + (size_t)start * CO);
        const unsigned laddr = (unsigned)(uintptr_t)(void*)sX
                             + (unsigned)(start - (b0 - 1)) * (CO * 2);
        // D# group0: count=1 | lds_addr | global_addr(57b) | type=2
        u32x4 g0;
        g0[0] = 1u;
        g0[1] = laddr;
        g0[2] = (unsigned)gaddr;
        g0[3] = (unsigned)((gaddr >> 32) & 0x1FFFFFFu) | (2u << 30);
        // D# group1: 2D tile, data_size=2B, row len 256, nrows rows, stride 256
        u32x8 g1;
        g1[0] = (1u << 16);          // workgroup_mask=0, data_size=1 (2 bytes)
        g1[1] = (256u << 16);        // tensor_dim0 = 256 elems
        g1[2] = (nrows << 16);       // tensor_dim1 = nrows
        g1[3] = (256u << 16);        // tile_dim0 = 256
        g1[4] = nrows;               // tile_dim1 = nrows, tile_dim2 = 0
        g1[5] = 256u;                // tensor_dim0_stride = 256 elems
        g1[6] = 0u;
        g1[7] = 0u;
        asm volatile("tensor_load_to_lds %0, %1" :: "s"(g0), "s"(g1) : "memory");
        __builtin_amdgcn_s_wait_tensorcnt(0);     // TENSORcnt == 0 -> tile in LDS
    }
    __syncthreads();

    const int lane = t & 31, wave = t >> 5;
    const int half = lane >> 4, ln = lane & 15;
    const int nt0 = wave * 2;
    const int co0 = nt0 * 16 + ln;
    const int co1 = co0 + 16;

    v8f acc[8];                                // acc[mt*2 + p]
    for (int j = 0; j < 8; ++j) for (int r = 0; r < 8; ++r) acc[j][r] = 0.f;

    for (int ks = 0; ks < 3; ++ks) {                 // kernel taps (b-1,b,b+1)
        const bf16* wk = wWk + ks * (8 * CO * 32);
        // warm L2/near cache for this k-slice (global_prefetch_b8)
        for (int pf = t; pf < 1024; pf += 256)
            __builtin_prefetch((const char*)wk + pf * 128, 0, 3);
        for (int kb = 0; kb < 8; ++kb) {             // ci blocks of 32
            const v16bf bm0 = *(const v16bf*)(wk + (kb << 13) + (co0 << 5) + (half << 4));
            const v16bf bm1 = *(const v16bf*)(wk + (kb << 13) + (co1 << 5) + (half << 4));
            #pragma unroll
            for (int mt = 0; mt < 4; ++mt) {
                v16bf a = load_a_frag(sX + (mt * 16 + ln + ks) * CO + kb * 32, half);
                acc[mt * 2 + 0] = wmma_bf16(a, bm0, acc[mt * 2 + 0]);
                acc[mt * 2 + 1] = wmma_bf16(a, bm1, acc[mt * 2 + 1]);
            }
        }
    }
    const float bias0 = b_conv[co0];
    const float bias1 = b_conv[co1];
    #pragma unroll
    for (int mt = 0; mt < 4; ++mt) {
        #pragma unroll
        for (int r = 0; r < 8; ++r) {
            const int gb = b0 + mt * 16 + half * 8 + r;
            const size_t base = (size_t)gb * (NN * CO) + node * CO;
            out[base + co0] = acc[mt * 2 + 0][r] + bias0;
            out[base + co1] = acc[mt * 2 + 1][r] + bias1;
        }
    }
}

// --- BatchNorm: deterministic 2-stage per-channel reduction + fused apply --
__global__ __launch_bounds__(256) void k_reduce1(const float* __restrict__ out,
                                                 float* __restrict__ ps,
                                                 float* __restrict__ pq) {
    const int t = threadIdx.x;
    const size_t base = (size_t)blockIdx.x * 16384;   // 16384 % 256 == 0 -> channel == t
    float s = 0.f, q = 0.f;
    for (int j = 0; j < 64; ++j) {
        const float v = out[base + (size_t)j * 256 + t];
        s += v; q += v * v;
    }
    ps[(size_t)blockIdx.x * 256 + t] = s;
    pq[(size_t)blockIdx.x * 256 + t] = q;
}

__global__ __launch_bounds__(256) void k_reduce2(const float* __restrict__ ps,
                                                 const float* __restrict__ pq,
                                                 const float* __restrict__ gamma,
                                                 const float* __restrict__ beta,
                                                 float* __restrict__ bnp) {
    const int co = threadIdx.x;
    float s = 0.f, q = 0.f;
    for (int j = 0; j < 2048; ++j) { s += ps[(size_t)j * 256 + co]; q += pq[(size_t)j * 256 + co]; }
    const float inv_n = 1.0f / (float)(BQ * NN);
    const float mean = s * inv_n;
    const float var = q * inv_n - mean * mean;
    const float sc = gamma[co] * rsqrtf(var + EPSV);
    bnp[co] = sc;
    bnp[256 + co] = beta[co] - mean * sc;
}

__global__ __launch_bounds__(256) void k_bnrelu(float* __restrict__ out,
                                                const float* __restrict__ bnp) {
    __shared__ float ssc[256], ssh[256];
    const int t = threadIdx.x;
    ssc[t] = bnp[t]; ssh[t] = bnp[256 + t];
    __syncthreads();
    const size_t idx = ((size_t)blockIdx.x * 256 + t) * 4;
    float4 v = *(const float4*)(out + idx);
    const int c0 = (int)(idx & 255);
    v.x = fmaxf(v.x * ssc[c0 + 0] + ssh[c0 + 0], 0.f);
    v.y = fmaxf(v.y * ssc[c0 + 1] + ssh[c0 + 1], 0.f);
    v.z = fmaxf(v.z * ssc[c0 + 2] + ssh[c0 + 2], 0.f);
    v.w = fmaxf(v.w * ssc[c0 + 3] + ssh[c0 + 3], 0.f);
    *(float4*)(out + idx) = v;
}

// --- host -------------------------------------------------------------------
extern "C" void kernel_launch(void* const* d_in, const int* in_sizes, int n_in,
                              void* d_out, int out_size, void* d_ws, size_t ws_size,
                              hipStream_t stream) {
    (void)in_sizes; (void)n_in; (void)out_size; (void)ws_size;
    const float* x      = (const float*)d_in[0];
    const float* W      = (const float*)d_in[1];
    const float* b_gcn  = (const float*)d_in[2];
    const float* w_conv = (const float*)d_in[3];
    const float* b_conv = (const float*)d_in[4];
    const float* gamma  = (const float*)d_in[5];
    const float* beta   = (const float*)d_in[6];
    const int*   ei     = (const int*)d_in[7];
    float* out = (float*)d_out;

    char* ws = (char*)d_ws;
    bf16* wA  = (bf16*)(ws + 0);                       //   8 KB
    bf16* wW  = (bf16*)(ws + 8192);                    //  64 KB
    bf16* wWk = (bf16*)(ws + 8192 + 65536);            // 384 KB
    bf16* xg  = (bf16*)(ws + 466944);                  //  64 MB bf16 x_gcn
    float* ps = (float*)(ws + 466944 + 67108864);      //   2 MB partial sums
    float* pq = ps + 2048 * 256;                       //   2 MB partial sumsq
    float* bnp = pq + 2048 * 256;                      //   2 KB scale/shift

    k_setup<<<1, 256, 0, stream>>>(W, w_conv, ei, wA, wW, wWk);
    k_gcn<<<BQ, 256, 0, stream>>>(x, b_gcn, wA, wW, xg);
    k_conv<<<dim3(BQ / 64, NN), 256, 0, stream>>>(xg, wWk, b_conv, out);
    k_reduce1<<<2048, 256, 0, stream>>>(out, ps, pq);
    k_reduce2<<<1, 256, 0, stream>>>(ps, pq, gamma, beta, bnp);
    k_bnrelu<<<(BQ * NN * CO) / (256 * 4), 256, 0, stream>>>(out, bnp);
}
